// GQAttention_43525198578341
// MI455X (gfx1250) — compile-verified, matrix operand order checked by reading, not run on hardware
//
#include <hip/hip_runtime.h>
#include <hip/hip_bf16.h>

// ---------------- problem constants (match reference) ----------------
constexpr int Bc       = 2;
constexpr int Tc       = 2048;
constexpr int Cc       = 2048;
constexpr int N_HEAD   = 16;
constexpr int N_KV     = 4;
constexpr int HEAD_DIM = Cc / N_HEAD;   // 128
constexpr int N_REP    = N_HEAD / N_KV; // 4
constexpr int HALF     = HEAD_DIM / 2;  // 64
constexpr float SCALE  = 0.08838834764831845f; // 1/sqrt(128)

typedef __attribute__((ext_vector_type(16))) __bf16 bf16x16;
typedef __attribute__((ext_vector_type(8)))  float  f32x8;

// ---------------- WMMA helpers ----------------
__device__ __forceinline__ f32x8 wmma_bf16(bf16x16 a, bf16x16 b, f32x8 c) {
    // (neg_a, A, neg_b, B, c_mod, C, reuse_a, reuse_b)
    return __builtin_amdgcn_wmma_f32_16x16x32_bf16(false, a, false, b,
                                                   (short)0, c, false, false);
}

// K index for element e (0..15) of a 16x32 bf16 A/B fragment (ISA 7.12.2):
// VGPR v=e/2: lanes 0-15 hold K = {0..7,16..23}, lanes 16-31 hold +8.
__device__ __forceinline__ int frag_k(int lane, int e) {
    int v = e >> 1;
    return ((lane >> 4) << 3) + ((v & 3) << 1) + ((v >> 2) << 4) + (e & 1);
}

// A-fragment (or B-fragment of a transposed-stored matrix):
// reads src[(m0 + lane%16)*ld + k0 + frag_k]
__device__ __forceinline__ bf16x16
load_rowmajor_frag(const __bf16* src, int ld, int m0, int k0, int lane) {
    const long row = m0 + (lane & 15);
    bf16x16 f;
#pragma unroll
    for (int e = 0; e < 16; ++e)
        f[e] = src[row * (long)ld + k0 + frag_k(lane, e)];
    return f;
}

// B-fragment from a row-major K x N matrix: reads src[(k0+frag_k)*ld + n0 + lane%16]
__device__ __forceinline__ bf16x16
load_colmajor_frag(const __bf16* src, int ld, int k0, int n0, int lane) {
    const int col = n0 + (lane & 15);
    bf16x16 f;
#pragma unroll
    for (int e = 0; e < 16; ++e)
        f[e] = src[(long)(k0 + frag_k(lane, e)) * ld + col];
    return f;
}

// ---------------- elementwise convert ----------------
__global__ void conv_f32_bf16(const float* __restrict__ in,
                              __bf16* __restrict__ out, long n) {
    long i = (long)blockIdx.x * blockDim.x + threadIdx.x;
    if (i < n) out[i] = (__bf16)in[i];
}

// ---------------- generic GEMM: C(MxN,f32) = A(MxK,bf16) * W(NxK,bf16)^T ----
// one wave computes a 32(M) x 64(N) tile; 8 waves per block.
// 8 WMMAs per k-step against 6 fragment loads (B fragments reused across M).
__global__ void gemm_abT(const __bf16* __restrict__ A,
                         const __bf16* __restrict__ W,
                         float* __restrict__ C,
                         int M, int N, int K) {
    const int lane = threadIdx.x & 31;
    const int wave = threadIdx.x >> 5;
    const long tile = (long)blockIdx.x * (blockDim.x >> 5) + wave;
    const int ntn = N >> 6;                      // N / 64
    const long ntiles = ((long)M >> 5) * ntn;    // M / 32
    if (tile >= ntiles) return;
    const int m0 = (int)(tile / ntn) << 5;
    const int n0 = (int)(tile % ntn) << 6;

    f32x8 acc[2][4] = {};
    for (int k0 = 0; k0 < K; k0 += 32) {
        // prefetch next k-slice (emits global_prefetch_b8, no counters)
        if (k0 + 32 < K) {
            __builtin_prefetch(&A[(long)(m0 + (lane & 15)) * K + k0 + 32], 0, 0);
            __builtin_prefetch(&W[(long)(n0 + lane) * K + k0 + 32], 0, 0);
        }
        bf16x16 a0 = load_rowmajor_frag(A, K, m0,      k0, lane);
        bf16x16 a1 = load_rowmajor_frag(A, K, m0 + 16, k0, lane);
#pragma unroll
        for (int j = 0; j < 4; ++j) {
            bf16x16 b = load_rowmajor_frag(W, K, n0 + j * 16, k0, lane);
            acc[0][j] = wmma_bf16(a0, b, acc[0][j]);
            acc[1][j] = wmma_bf16(a1, b, acc[1][j]);
        }
    }
    const int col = lane & 15;
#pragma unroll
    for (int i = 0; i < 2; ++i) {
        const int rbase = m0 + i * 16 + ((lane >> 4) << 3);
#pragma unroll
        for (int j = 0; j < 4; ++j)
#pragma unroll
            for (int v = 0; v < 8; ++v)
                C[(long)(rbase + v) * N + n0 + j * 16 + col] = acc[i][j][v];
    }
}

// ---------------- RoPE + repack: proj(B*T, nh*D) f32 -> out[B][nh][T][D] bf16
__global__ void rope_pack(const float* __restrict__ proj,
                          const float* __restrict__ cosT,
                          const float* __restrict__ sinT,
                          __bf16* __restrict__ out, int nheads) {
    long idx = (long)blockIdx.x * blockDim.x + threadIdx.x;
    const long total = (long)Bc * Tc * nheads * HALF;
    if (idx >= total) return;
    const int i = idx % HALF;  idx /= HALF;
    const int h = idx % nheads; idx /= nheads;
    const int t = idx % Tc;
    const int b = idx / Tc;
    const long row = ((long)b * Tc + t) * (nheads * HEAD_DIM) + h * HEAD_DIM;
    const float e = proj[row + 2 * i];
    const float o = proj[row + 2 * i + 1];
    const float c = cosT[(long)t * HALF + i];
    const float s = sinT[(long)t * HALF + i];
    const long ob = (((long)b * nheads + h) * Tc + t) * HEAD_DIM;
    out[ob + 2 * i]     = (__bf16)(e * c - o * s);
    out[ob + 2 * i + 1] = (__bf16)(e * s + o * c);
}

// ---------------- V repack: proj(B*T, KV*D) f32 -> out[B][KV][T][D] bf16 ----
__global__ void pack_v(const float* __restrict__ proj, __bf16* __restrict__ out) {
    long idx = (long)blockIdx.x * blockDim.x + threadIdx.x;
    const long total = (long)Bc * Tc * N_KV * HEAD_DIM;
    if (idx >= total) return;
    const int d = idx % HEAD_DIM; idx /= HEAD_DIM;
    const int g = idx % N_KV;     idx /= N_KV;
    const int t = idx % Tc;
    const int b = idx / Tc;
    out[(((long)b * N_KV + g) * Tc + t) * HEAD_DIM + d] =
        (__bf16)proj[((long)b * Tc + t) * (N_KV * HEAD_DIM) + g * HEAD_DIM + d];
}

// ---------------- flash attention: one wave per (b, head, 16-query tile) ----
// Q[B][H][T][D], K/V[B][KV][T][D] bf16 -> Y (B*T, C) bf16
__global__ void attn_fwd(const __bf16* __restrict__ Q,
                         const __bf16* __restrict__ Kt,
                         const __bf16* __restrict__ V,
                         __bf16* __restrict__ Y) {
    __shared__ float  sS[16 * 32];
    __shared__ __bf16 sP[16 * 32];
    __shared__ float  sM[16], sL[16], sA[16];

    const int lane = threadIdx.x;
    const int t0 = blockIdx.x << 4;
    const int h  = blockIdx.y;
    const int b  = blockIdx.z;
    const int g  = h / N_REP;

    const __bf16* Qb = Q  + (((long)b * N_HEAD + h) * Tc + t0) * HEAD_DIM;
    const __bf16* Kb = Kt + (((long)b * N_KV  + g) * Tc) * HEAD_DIM;
    const __bf16* Vb = V  + (((long)b * N_KV  + g) * Tc) * HEAD_DIM;

    bf16x16 qf[4];
#pragma unroll
    for (int c = 0; c < 4; ++c)
        qf[c] = load_rowmajor_frag(Qb, HEAD_DIM, 0, c * 32, lane);

    f32x8 o[8] = {};
    if (lane < 16) { sM[lane] = -1e30f; sL[lane] = 0.0f; }
    __syncthreads();

    const int grp = (lane >> 4) << 3;   // accumulator row base within tile
    const int col = lane & 15;
    const int send = t0 + 16;           // keys processed: s < send (masked per row)

    for (int s0 = 0; s0 < send; s0 += 32) {
        // prefetch the K/V rows of the next key step
        if (s0 + 32 < send) {
            __builtin_prefetch(&Kb[(long)(s0 + 32 + lane) * HEAD_DIM], 0, 0);
            __builtin_prefetch(&Vb[(long)(s0 + 32 + lane) * HEAD_DIM], 0, 0);
        }
        // ---- S = scale * Q @ K^T  (two 16x16 tiles -> LDS, causal-masked) ----
#pragma unroll
        for (int half = 0; half < 2; ++half) {
            const int sb = s0 + half * 16;
            f32x8 acc = {};
            if (sb < send) {
#pragma unroll
                for (int c = 0; c < 4; ++c) {
                    bf16x16 kf = load_rowmajor_frag(Kb, HEAD_DIM, sb, c * 32, lane);
                    acc = wmma_bf16(qf[c], kf, acc);
                }
            }
#pragma unroll
            for (int v = 0; v < 8; ++v) {
                const int row = grp + v;
                const int s = sb + col;
                const float val = (sb < send && s <= t0 + row)
                                      ? acc[v] * SCALE : -1e30f;
                sS[row * 32 + half * 16 + col] = val;
            }
        }
        __syncthreads();

        // ---- online softmax: lanes 0..15 own one query row each ----
        if (lane < 16) {
            const int row = lane;
            float mNew = sM[row];
#pragma unroll 4
            for (int j = 0; j < 32; ++j) mNew = fmaxf(mNew, sS[row * 32 + j]);
            const float alpha = __expf(sM[row] - mNew);
            float lsum = 0.0f;
#pragma unroll 4
            for (int j = 0; j < 32; ++j) {
                const float p = __expf(sS[row * 32 + j] - mNew);
                sP[row * 32 + j] = (__bf16)p;
                lsum += p;
            }
            sM[row] = mNew;
            sL[row] = sL[row] * alpha + lsum;
            sA[row] = alpha;
        }
        __syncthreads();

        // ---- rescale accumulators, then O += P @ V ----
        float al[8];
#pragma unroll
        for (int v = 0; v < 8; ++v) al[v] = sA[grp + v];

        bf16x16 pf = load_rowmajor_frag(sP, 32, 0, 0, lane);
#pragma unroll
        for (int j = 0; j < 8; ++j) {
#pragma unroll
            for (int v = 0; v < 8; ++v) o[j][v] *= al[v];
            bf16x16 vf = load_colmajor_frag(Vb, HEAD_DIM, s0, j * 16, lane);
            o[j] = wmma_bf16(pf, vf, o[j]);
        }
        __syncthreads();
    }

    // ---- normalize and store Y (B*T, C) bf16 ----
    float li[8];
#pragma unroll
    for (int v = 0; v < 8; ++v) li[v] = 1.0f / sL[grp + v];
#pragma unroll
    for (int j = 0; j < 8; ++j)
#pragma unroll
        for (int v = 0; v < 8; ++v) {
            const long t = t0 + grp + v;
            Y[((long)b * Tc + t) * Cc + h * HEAD_DIM + j * 16 + col] =
                (__bf16)(o[j][v] * li[v]);
        }
}

// ---------------- host launch ----------------
extern "C" void kernel_launch(void* const* d_in, const int* in_sizes, int n_in,
                              void* d_out, int out_size, void* d_ws, size_t ws_size,
                              hipStream_t stream) {
    const float* x    = (const float*)d_in[0];
    const float* fcos = (const float*)d_in[1];
    const float* fsin = (const float*)d_in[2];
    const float* wq   = (const float*)d_in[3];
    const float* wk   = (const float*)d_in[4];
    const float* wv   = (const float*)d_in[5];
    const float* wo   = (const float*)d_in[6];

    char* ws = (char*)d_ws;
    size_t off = 0;
    auto alloc = [&](size_t bytes) -> void* {
        void* p = ws + off;
        off += (bytes + 255) & ~(size_t)255;
        return p;
    };

    const long MT  = (long)Bc * Tc;          // 4096 rows
    const long KVC = (long)N_KV * HEAD_DIM;  // 512

    __bf16* xb   = (__bf16*)alloc(MT * Cc * sizeof(__bf16));
    __bf16* wqb  = (__bf16*)alloc((size_t)Cc * Cc * sizeof(__bf16));
    __bf16* wkb  = (__bf16*)alloc((size_t)KVC * Cc * sizeof(__bf16));
    __bf16* wvb  = (__bf16*)alloc((size_t)KVC * Cc * sizeof(__bf16));
    __bf16* wob  = (__bf16*)alloc((size_t)Cc * Cc * sizeof(__bf16));
    float*  prj  = (float*) alloc(MT * Cc * sizeof(float));     // reused q->k->v
    __bf16* Qb   = (__bf16*)alloc(MT * Cc * sizeof(__bf16));
    __bf16* Kb   = (__bf16*)alloc(MT * KVC * sizeof(__bf16));
    __bf16* Vb   = (__bf16*)alloc(MT * KVC * sizeof(__bf16));
    __bf16* Yb   = (__bf16*)alloc(MT * Cc * sizeof(__bf16));

    auto conv = [&](const float* src, __bf16* dst, long n) {
        conv_f32_bf16<<<(int)((n + 255) / 256), 256, 0, stream>>>(src, dst, n);
    };
    conv(x,  xb,  MT * Cc);
    conv(wq, wqb, (long)Cc * Cc);
    conv(wk, wkb, KVC * Cc);
    conv(wv, wvb, KVC * Cc);
    conv(wo, wob, (long)Cc * Cc);

    auto gemm = [&](const __bf16* A, const __bf16* W, float* C,
                    int M, int N, int K) {
        long tiles = ((long)M / 32) * (N / 64);
        gemm_abT<<<(int)((tiles + 7) / 8), 256, 0, stream>>>(A, W, C, M, N, K);
    };

    // Q projection + RoPE
    gemm(xb, wqb, prj, (int)MT, Cc, Cc);
    {
        long n = MT * N_HEAD * HALF;
        rope_pack<<<(int)((n + 255) / 256), 256, 0, stream>>>(prj, fcos, fsin, Qb, N_HEAD);
    }
    // K projection + RoPE
    gemm(xb, wkb, prj, (int)MT, (int)KVC, Cc);
    {
        long n = MT * N_KV * HALF;
        rope_pack<<<(int)((n + 255) / 256), 256, 0, stream>>>(prj, fcos, fsin, Kb, N_KV);
    }
    // V projection + pack
    gemm(xb, wvb, prj, (int)MT, (int)KVC, Cc);
    {
        long n = MT * KVC;
        pack_v<<<(int)((n + 255) / 256), 256, 0, stream>>>(prj, Vb);
    }

    // attention
    attn_fwd<<<dim3(Tc / 16, N_HEAD, Bc), 32, 0, stream>>>(Qb, Kb, Vb, Yb);

    // output projection -> fp32 d_out
    gemm(Yb, wob, (float*)d_out, (int)MT, Cc, Cc);
    (void)in_sizes; (void)n_in; (void)out_size; (void)ws_size;
}
